// SelfAttention_2534030704940
// MI455X (gfx1250) — matrix operand order repaired and checked by reading, not compiled, not run on hardware
//
#include <hip/hip_runtime.h>

// Chunked self-attention for MI455X (gfx1250, wave32, WMMA, TDM).
// One workgroup per (batch, chunk). x-slab DMA'd to LDS via tensor_load_to_lds,
// all GEMMs via v_wmma_f32_16x16x32_f16 with LDS layouts chosen so every
// fragment load is two ds_load_b128 and every tile store is contiguous.
// HBM traffic = read x + write out (roofline-bound at 23.3 TB/s).

typedef __attribute__((ext_vector_type(16))) _Float16 v16h;
typedef __attribute__((ext_vector_type(8)))  _Float16 v8h;
typedef __attribute__((ext_vector_type(8)))  float    v8f;
typedef __attribute__((ext_vector_type(4)))  float    v4f;
typedef __attribute__((ext_vector_type(4)))  unsigned int v4u;
typedef __attribute__((ext_vector_type(8)))  int      v8i;
typedef __attribute__((ext_vector_type(4)))  int      v4i;

namespace {
constexpr int B    = 8;
constexpr int C    = 64;
constexpr int T    = 100000;
constexpr int CH   = 100;       // chunk length
constexpr int NCH  = T / CH;    // 1000 chunks
constexpr int MP   = 112;       // chunk padded to 7*16
constexpr int NT   = MP / 16;   // 7 time tiles
constexpr int KP   = 128;       // contraction dim padded to 4*32
constexpr int DQ   = 8;         // C/8
constexpr int DP   = 16;        // DQ padded (WMMA K padded to 32 via zeros)

// LDS strides in halves/floats -- all multiples of 8 halves for b128 alignment
constexpr int SW  = C + 8;      // 72  : weight / X_cm row stride (halves)
constexpr int SD  = 32;         // Qt/Kt row stride (halves), d padded to 32
constexpr int SVS = KP + 8;     // 136 : V row stride (halves)
constexpr int SAS = KP + 8;     // 136 : A_cm row stride (halves)
constexpr int SES = MP + 4;     // 116 : energy row stride (floats)
}

union AFr { v16h v; v8h p[2]; _Float16 h[16]; };
union CFr { v8f  v; float f[8]; };
union HF  { v8h  v; _Float16 h[8]; };
union FF  { v4f  v; float f[4]; };

__device__ __forceinline__ v8h zero8h() {
  HF o;
#pragma unroll
  for (int j = 0; j < 8; ++j) o.h[j] = (_Float16)0.f;
  return o.v;
}

// A-fragment (16x32, MxK) from row-major [M][K] storage: two contiguous b128s.
// ISA: lanes0-15 m, v0-3:K(g*8+0..7), v4-7:K(16+g*8+0..7); lanes16-31 mirror.
__device__ __forceinline__ v16h load_a_rm(const _Float16* base, int stride, int row0, int k0, int lane) {
  const int m = lane & 15, g = lane >> 4;
  const _Float16* p = base + (row0 + m) * stride + k0 + g * 8;
  AFr f;
  f.p[0] = *(const v8h*)(p);
  f.p[1] = *(const v8h*)(p + 16);
  return f.v;
}

// B-fragment (32x16, KxN) from column-major [N][K] storage: two contiguous b128s.
// ISA: lanes0-15 n, v0-3:K(g*8+0..7), v4-7:K(16+g*8+0..7); lanes16-31 mirror (matches A).
__device__ __forceinline__ v16h load_b_cm(const _Float16* base, int stride, int k0, int col0, int lane) {
  const int n = lane & 15, g = lane >> 4;
  const _Float16* p = base + (col0 + n) * stride + k0 + g * 8;
  AFr f;
  f.p[0] = *(const v8h*)(p);
  f.p[1] = *(const v8h*)(p + 16);
  return f.v;
}

__global__ __launch_bounds__(256)
void chunked_attn_kernel(const float* __restrict__ x,
                         const float* __restrict__ Wq, const float* __restrict__ bq,
                         const float* __restrict__ Wk, const float* __restrict__ bk,
                         const float* __restrict__ Wv, const float* __restrict__ bv,
                         const float* __restrict__ gamma,
                         float* __restrict__ out)
{
  // Staging (X_cm + weights) dead after phase 1 -> overlapped with A_cm.
  struct Stage {
    _Float16 X [MP][SW];   // X_cm[t][cin] (col-major vs GEMM use); rows 100..111 zero
    _Float16 Wq[DP][SW];   // row-major [d][cin], d 8..15 zero
    _Float16 Wk[DP][SW];
    _Float16 Wv[C][SW];    // row-major [cout][cin]
  };
  __shared__ alignas(16) union { Stage st; _Float16 A[MP][SAS]; } u;   // A_cm[k][s]
  // f32 x staging (TDM target) dead before energy -> overlapped with sE.
  __shared__ alignas(16) union { float E[MP][SES]; float Xs[C * CH]; } eu;
  __shared__ alignas(16) _Float16 sQt[MP][SD];   // Q^T[t][d], d 16..31 zero
  __shared__ alignas(16) _Float16 sKt[MP][SD];   // K^T[t][d]
  __shared__ alignas(16) _Float16 sV [C][SVS];   // V[c][s], s 112..127 zero
  __shared__ float rmax[MP], rinv[MP];
  __shared__ float sBq[DP], sBk[DP], sBv[C];
  __shared__ float sG;

  const int tid  = threadIdx.x;
  const int wave = tid >> 5;
  const int lane = tid & 31;
  const int nn   = lane & 15;
  const int g    = lane >> 4;

  const int b  = blockIdx.x / NCH;
  const int n  = blockIdx.x % NCH;
  const int t0 = n * CH;
  const float* xb = x   + (size_t)b * C * T + t0;
  float*       ob = out + (size_t)b * C * T + t0;

  // ---------------- phase 0a: TDM the x-slab (C x CH f32, row stride T) into LDS ----------------
  if (wave == 0) {
    const unsigned long long ga = (unsigned long long)(size_t)xb;
    const unsigned int lds_off  = (unsigned int)(size_t)(&eu.Xs[0]);
    v4u g0;
    g0.x = 1u;                                               // count = 1 valid descriptor
    g0.y = lds_off;                                          // lds_addr (bytes)
    g0.z = (unsigned int)(ga & 0xffffffffu);                 // global_addr[31:0]
    g0.w = (unsigned int)((ga >> 32) & 0x01ffffffu) | (2u << 30);  // addr[56:32] | type=2
    v8i g1;
    g1[0] = (int)(2u << 16);                 // workgroup_mask=0, data_size=4B
    g1[1] = (int)(((unsigned)CH) << 16);     // tensor_dim0[15:0] = 100
    g1[2] = (int)(((unsigned)C)  << 16);     // tensor_dim1[15:0] = 64
    g1[3] = (int)(((unsigned)CH) << 16);     // tile_dim0 = 100 (400B, mult of 4B)
    g1[4] = (int)C;                          // tile_dim1 = 64, tile_dim2 = 0
    g1[5] = (int)T;                          // tensor_dim0_stride[31:0] = 100000
    g1[6] = 0;
    g1[7] = 0;
    v4i gz4  = {0, 0, 0, 0};                 // 2D tensor: groups 2/3 unused
    v8i gz8  = {0, 0, 0, 0, 0, 0, 0, 0};     // trailing group (clang-23 6-arg form)
    __builtin_amdgcn_tensor_load_to_lds(g0, g1, gz4, gz4, gz8, 0);
  }

  // ---------------- phase 0b: stage weights/biases, zero pads (overlaps TDM) ----------------
  if (tid == 0) sG = gamma[0];
  for (int i = tid; i < DP; i += 256) {
    sBq[i] = (i < DQ) ? bq[i] : 0.f;
    sBk[i] = (i < DQ) ? bk[i] : 0.f;
  }
  for (int i = tid; i < C; i += 256) sBv[i] = bv[i];
  for (int i = tid; i < DP * C; i += 256) {
    int r = i >> 6, c = i & 63;
    u.st.Wq[r][c] = (r < DQ) ? (_Float16)Wq[r * C + c] : (_Float16)0.f;
    u.st.Wk[r][c] = (r < DQ) ? (_Float16)Wk[r * C + c] : (_Float16)0.f;
  }
  for (int i = tid; i < C * C; i += 256) {
    int r = i >> 6, c = i & 63;
    u.st.Wv[r][c] = (_Float16)Wv[i];
  }
  {
    const v8h z8 = zero8h();
    for (int i = tid; i < (C * SVS) / 8; i += 256)             // zero V incl. s>=100 pad
      ((v8h*)&sV[0][0])[i] = z8;
    for (int i = tid; i < ((MP - CH) * SW) / 8; i += 256)      // zero X_cm rows t=100..111
      ((v8h*)&u.st.X[CH][0])[i] = z8;
  }
  if (wave == 0) __builtin_amdgcn_s_wait_tensorcnt(0);
  __syncthreads();

  // ---------------- phase 0c: transpose-convert x f32 -> X_cm[t][cin] f16 ----------------
  for (int i = tid; i < CH * C; i += 256) {
    int t = i >> 6, c = i & 63;
    u.st.X[t][c] = (_Float16)eu.Xs[c * CH + t];
  }
  __syncthreads();

  // ---------------- phase 1: Q^T, K^T, V (all stores contiguous b128) ----------------
  // jobs: [0,7) Q, [7,14) K, [14,42) V^T (7 s-tiles x 4 c-tiles)
  for (int job = wave; job < 2 * NT + 4 * NT; job += 8) {
    CFr acc;
#pragma unroll
    for (int r = 0; r < 8; ++r) acc.f[r] = 0.f;

    if (job < 2 * NT) {                              // Q or K tile: D[d][t] -> store [t][d]
      const bool isQ = job < NT;
      const int  nt  = isQ ? job : job - NT;
      const _Float16* W = isQ ? &u.st.Wq[0][0] : &u.st.Wk[0][0];
#pragma unroll
      for (int kk = 0; kk < C; kk += 32) {
        v16h a  = load_a_rm(W, SW, 0, kk, lane);
        v16h bf = load_b_cm(&u.st.X[0][0], SW, kk, nt * 16, lane);
        acc.v = __builtin_amdgcn_wmma_f32_16x16x32_f16(false, a, false, bf, (short)0, acc.v, false, false);
      }
      const int t = nt * 16 + nn;
      const float* bias = isQ ? sBq : sBk;
      _Float16* dst = isQ ? &sQt[t][0] : &sKt[t][0];
      HF o;
#pragma unroll
      for (int r = 0; r < 8; ++r) o.h[r] = (_Float16)(acc.f[r] + bias[8 * g + r]);
      *(v8h*)(dst + 8 * g)      = o.v;         // d = 8g..8g+7
      *(v8h*)(dst + 16 + 8 * g) = zero8h();    // d = 16..31 zero pad for k=32 WMMA
    } else {                                         // V^T tile: D[s][c] -> store V[c][s]
      const int v  = job - 2 * NT;
      const int st = v >> 2, ct = v & 3;
#pragma unroll
      for (int kk = 0; kk < C; kk += 32) {
        v16h a  = load_a_rm(&u.st.X[0][0], SW, st * 16, kk, lane);   // A[s][cin]
        v16h bf = load_b_cm(&u.st.Wv[0][0], SW, kk, ct * 16, lane);  // B[cin][cout]
        acc.v = __builtin_amdgcn_wmma_f32_16x16x32_f16(false, a, false, bf, (short)0, acc.v, false, false);
      }
      const int c   = ct * 16 + nn;
      const float bvv = sBv[c];
      HF o;
#pragma unroll
      for (int r = 0; r < 8; ++r) o.h[r] = (_Float16)(acc.f[r] + bvv);
      *(v8h*)&sV[c][st * 16 + 8 * g] = o.v;    // 8 consecutive s at fixed c
    }
  }
  __syncthreads();

  // ---------------- phase 2: e^T = Kt x Qt^T, stored natural sE[q][k] (contiguous) ----------------
  for (int job = wave; job < NT * NT; job += 8) {
    const int kt = job / NT, qt = job % NT;
    CFr acc;
#pragma unroll
    for (int r = 0; r < 8; ++r) acc.f[r] = 0.f;
    v16h a  = load_a_rm(&sKt[0][0], SD, kt * 16, 0, lane);    // A[k][d]
    v16h bf = load_b_cm(&sQt[0][0], SD, 0, qt * 16, lane);    // B[d][q]
    acc.v = __builtin_amdgcn_wmma_f32_16x16x32_f16(false, a, false, bf, (short)0, acc.v, false, false);
    const int q  = qt * 16 + nn;               // lane col = q
    const int kb = kt * 16 + 8 * g;            // 8 consecutive k
    FF a0, a1;
#pragma unroll
    for (int j = 0; j < 4; ++j) { a0.f[j] = acc.f[j]; a1.f[j] = acc.f[4 + j]; }
    *(v4f*)&eu.E[q][kb]     = a0.v;
    *(v4f*)&eu.E[q][kb + 4] = a1.v;
  }
  __syncthreads();

  // ---------------- phase 3a: per-row softmax stats (contiguous row reads) ----------------
  for (int q = tid; q < CH; q += 256) {
    float mx = -3.4e38f;
    for (int k = 0; k < CH; ++k) mx = fmaxf(mx, eu.E[q][k]);
    float s = 0.f;
    for (int k = 0; k < CH; ++k) s += __expf(eu.E[q][k] - mx);
    rmax[q] = mx;
    rinv[q] = 1.f / s;
  }
  __syncthreads();

  // ---------------- phase 3b: emit A_cm[k][s] = A[s][k] as f16 (b128 stores) ----------------
  for (int idx = tid; idx < MP * (KP / 8); idx += 256) {
    const int k  = idx >> 4;
    const int s0 = (idx & 15) << 3;
    HF o;
#pragma unroll
    for (int j = 0; j < 8; ++j) {
      const int s = s0 + j;
      o.h[j] = (s < CH) ? (_Float16)(__expf(eu.E[s][k] - rmax[s]) * rinv[s]) : (_Float16)0.f;
    }
    *(v8h*)&u.A[k][s0] = o.v;
  }
  __syncthreads();

  // ---------------- phase 4: out^T = A^T x V^T, coalesced residual + store ----------------
  for (int job = wave; job < 4 * NT; job += 8) {
    const int tt = job >> 2, ct = job & 3;
    CFr acc;
#pragma unroll
    for (int r = 0; r < 8; ++r) acc.f[r] = 0.f;
#pragma unroll
    for (int kk = 0; kk < KP; kk += 32) {
      v16h a  = load_a_rm(&u.A[0][0], SAS, tt * 16, kk, lane);   // A^T[k][s]
      v16h bf = load_b_cm(&sV[0][0], SVS, kk, ct * 16, lane);    // B[s][c] = V[c][s]
      acc.v = __builtin_amdgcn_wmma_f32_16x16x32_f16(false, a, false, bf, (short)0, acc.v, false, false);
    }
    const int   c   = ct * 16 + nn;            // lane col = channel
    const int   tb  = tt * 16 + 8 * g;         // 8 consecutive t
    const float gmv = sG;
    const float* xr = xb + (size_t)c * T + tb;
    float*      orw = ob + (size_t)c * T + tb;
    if (tb + 8 <= CH) {                        // full vector: 2x float4 load + store
      v4f x0 = *(const v4f*)xr;
      v4f x1 = *(const v4f*)(xr + 4);
      FF o0, o1;
#pragma unroll
      for (int j = 0; j < 4; ++j) { o0.f[j] = gmv * acc.f[j] + x0[j]; o1.f[j] = gmv * acc.f[4 + j] + x1[j]; }
      *(v4f*)orw       = o0.v;
      *(v4f*)(orw + 4) = o1.v;
    } else if (tb < CH) {                      // boundary tile t=96..99
#pragma unroll
      for (int r = 0; r < 8; ++r)
        if (tb + r < CH) orw[r] = gmv * acc.f[r] + xr[r];
    }
  }
}

extern "C" void kernel_launch(void* const* d_in, const int* in_sizes, int n_in,
                              void* d_out, int out_size, void* d_ws, size_t ws_size,
                              hipStream_t stream) {
  (void)in_sizes; (void)n_in; (void)out_size; (void)d_ws; (void)ws_size;
  const float* x  = (const float*)d_in[0];
  const float* Wq = (const float*)d_in[1];
  const float* bq = (const float*)d_in[2];
  const float* Wk = (const float*)d_in[3];
  const float* bk = (const float*)d_in[4];
  const float* Wv = (const float*)d_in[5];
  const float* bv = (const float*)d_in[6];
  const float* gm = (const float*)d_in[7];
  chunked_attn_kernel<<<dim3(B * NCH), dim3(256), 0, stream>>>(
      x, Wq, bq, Wk, bk, Wv, bv, gm, (float*)d_out);
}
